// self_attention_12670153523745
// MI455X (gfx1250) — matrix operand order repaired and checked by reading, not compiled, not run on hardware
//
#include <hip/hip_runtime.h>

#define SEQ 4096
#define HID 768
#define NH  12
#define HD  64

typedef __bf16 v16bf __attribute__((ext_vector_type(16)));
typedef float  v8f   __attribute__((ext_vector_type(8)));

union BF8 { float4 f; __bf16 b[8]; };

__device__ __forceinline__ v8f wmma_bf16(v16bf a, v16bf b, v8f c) {
  // D = A(16x32 bf16) * B(32x16 bf16) + C(16x16 f32)
  return __builtin_amdgcn_wmma_f32_16x16x32_bf16(false, a, false, b, (short)0, c,
                                                 false, false);
}

// Build a fragment from two 16B chunks (elems 0-7 / 8-15).
__device__ __forceinline__ v16bf frag_bf16(const __bf16* p0, const __bf16* p1) {
  BF8 lo, hi;
  lo.f = *(const float4*)p0;
  hi.f = *(const float4*)p1;
  v16bf r;
#pragma unroll
  for (int i = 0; i < 8; ++i) { r[i] = lo.b[i]; r[i + 8] = hi.b[i]; }
  return r;
}

// A fragment sourced from fp32 memory (convert to bf16 on the fly).
__device__ __forceinline__ v16bf frag_f32cvt(const float* p0, const float* p1) {
  float4 a0 = *(const float4*)p0;
  float4 a1 = *(const float4*)(p0 + 4);
  float4 b0 = *(const float4*)p1;
  float4 b1 = *(const float4*)(p1 + 4);
  v16bf r;
  r[0] = (__bf16)a0.x; r[1] = (__bf16)a0.y; r[2]  = (__bf16)a0.z; r[3]  = (__bf16)a0.w;
  r[4] = (__bf16)a1.x; r[5] = (__bf16)a1.y; r[6]  = (__bf16)a1.z; r[7]  = (__bf16)a1.w;
  r[8] = (__bf16)b0.x; r[9] = (__bf16)b0.y; r[10] = (__bf16)b0.z; r[11] = (__bf16)b0.w;
  r[12] = (__bf16)b1.x; r[13] = (__bf16)b1.y; r[14] = (__bf16)b1.z; r[15] = (__bf16)b1.w;
  return r;
}

// ---- CDNA5 async global->LDS copy (ASYNCcnt path) via inline asm ----------
// global_load_async_to_lds_b128: VDST = per-lane LDS byte offset (low 32 bits
// of a generic pointer to __shared__), ADDR = 64-bit global address.
#define USE_ASYNC_LDS 1

__device__ __forceinline__ void async_copy16(const __bf16* g, __bf16* l) {
#if USE_ASYNC_LDS
  unsigned lds = (unsigned)(unsigned long long)l;
  asm volatile("global_load_async_to_lds_b128 %0, %1, off"
               :
               : "v"(lds), "v"(g)
               : "memory");
#else
  *(float4*)l = *(const float4*)g;
#endif
}

__device__ __forceinline__ void wait_async() {
#if USE_ASYNC_LDS
  asm volatile("s_wait_asynccnt 0x0" ::: "memory");
#endif
}

// ---------------------------------------------------------------------------
// Kernel 1: fused QKV projection. grid (16, 12, 3), block 256 (8 waves).
// Each wave computes a 32(M)x64(N) tile (2 A-frags reuse 4 B-frags = 8 WMMA).
// z=0 -> Q (x 1/8) as [h][S][64]; z=1 -> K as [h][S][64]; z=2 -> V^T [h][64][S].
// ---------------------------------------------------------------------------
__global__ __launch_bounds__(256) void qkv_proj_kernel(
    const float* __restrict__ x, const float* __restrict__ Wq,
    const float* __restrict__ Wk, const float* __restrict__ Wv,
    __bf16* __restrict__ Qbf, __bf16* __restrict__ Kbf,
    __bf16* __restrict__ Vtbf) {
  __shared__ __bf16 Bs[64 * 72];  // W tile transposed: [n_local][k], stride 72
  const int m0 = blockIdx.x * 256;
  const int n0 = blockIdx.y * 64;
  const int z  = blockIdx.z;
  const float* W = (z == 0) ? Wq : (z == 1) ? Wk : Wv;
  const int tid  = threadIdx.x;
  const int lane = tid & 31;
  const int wv   = tid >> 5;
  const int hl   = lane >> 4;
  const int l16  = lane & 15;
  const int mrow0 = m0 + wv * 32;

  v8f acc[2][4];
#pragma unroll
  for (int s = 0; s < 2; ++s)
#pragma unroll
    for (int i = 0; i < 4; ++i)
      acc[s][i] = (v8f){0.f, 0.f, 0.f, 0.f, 0.f, 0.f, 0.f, 0.f};

  for (int kk = 0; kk < HID; kk += 32) {
    __syncthreads();
#pragma unroll
    for (int i = 0; i < 2; ++i) {
      int idx = tid + i * 256;          // 512 float4 = 32x64 fp32 tile of W
      int k = idx >> 4, ng = idx & 15;
      float4 f = *(const float4*)&W[(size_t)(kk + k) * HID + n0 + ng * 4];
      Bs[(ng * 4 + 0) * 72 + k] = (__bf16)f.x;
      Bs[(ng * 4 + 1) * 72 + k] = (__bf16)f.y;
      Bs[(ng * 4 + 2) * 72 + k] = (__bf16)f.z;
      Bs[(ng * 4 + 3) * 72 + k] = (__bf16)f.w;
    }
    __syncthreads();
    v16bf bfr[4];  // preload all B fragments, then batch the WMMAs
#pragma unroll
    for (int nt = 0; nt < 4; ++nt) {
      const __bf16* bp = &Bs[(nt * 16 + l16) * 72 + 16 * hl];
      bfr[nt] = frag_bf16(bp, bp + 8);
    }
#pragma unroll
    for (int sub = 0; sub < 2; ++sub) {
      const float* xr = &x[(size_t)(mrow0 + sub * 16 + l16) * HID + kk];
      v16bf a = frag_f32cvt(xr + 16 * hl, xr + 16 * hl + 8);
#pragma unroll
      for (int nt = 0; nt < 4; ++nt)
        acc[sub][nt] = wmma_bf16(a, bfr[nt], acc[sub][nt]);
    }
  }

#pragma unroll
  for (int sub = 0; sub < 2; ++sub) {
#pragma unroll
    for (int nt = 0; nt < 4; ++nt) {
      int n = n0 + nt * 16 + l16;
      int h = n >> 6, d = n & 63;
      if (z == 2) {  // V^T: 8 consecutive seq rows -> one b128 store
        BF8 pk;
#pragma unroll
        for (int r = 0; r < 8; ++r) pk.b[r] = (__bf16)acc[sub][nt][r];
        *(float4*)&Vtbf[((size_t)(h * HD + d)) * SEQ + mrow0 + sub * 16 + 8 * hl] =
            pk.f;
      } else {
        __bf16* dst = (z == 0) ? Qbf : Kbf;
        const float scale = (z == 0) ? 0.125f : 1.0f;  // 1/sqrt(64) in Q
#pragma unroll
        for (int r = 0; r < 8; ++r) {
          int m = mrow0 + sub * 16 + r + 8 * hl;
          dst[((size_t)(h * SEQ + m)) * HD + d] = (__bf16)(acc[sub][nt][r] * scale);
        }
      }
    }
  }
}

// ---------------------------------------------------------------------------
// Kernel 2: flash attention. Block = 4 waves = 4 consecutive 16-query tiles of
// ONE head; K/V tiles (64 keys) double-buffered in LDS via async global->LDS.
// grid 768 blocks x 128 threads.
// ---------------------------------------------------------------------------
__global__ __launch_bounds__(128) void flash_attn_kernel(
    const __bf16* __restrict__ Qbf, const __bf16* __restrict__ Kbf,
    const __bf16* __restrict__ Vtbf, __bf16* __restrict__ Abf) {
  __shared__ __bf16 Ks[2][64 * 72];   // [t_local][d], stride 72
  __shared__ __bf16 Vs[2][64 * 72];   // [d][t_local], stride 72
  __shared__ __bf16 Pl[4][16 * 72];   // per-wave P tile
  const int tid  = threadIdx.x;
  const int lane = tid & 31;
  const int wv   = tid >> 5;
  const int hl   = lane >> 4;
  const int l16  = lane & 15;
  const int h    = blockIdx.x >> 6;          // 64 blocks per head
  const int q0   = (blockIdx.x & 63) * 64 + wv * 16;

  const __bf16* Qh = Qbf + (size_t)h * SEQ * HD;
  const __bf16* Kh = Kbf + (size_t)h * SEQ * HD;
  const __bf16* Vh = Vtbf + (size_t)h * HD * SEQ;
  __bf16* pl = &Pl[wv][0];

  // Q row fragments (d-chunks 0..31 / 32..63) stay in VGPRs for the whole loop
  v16bf qa[2];
#pragma unroll
  for (int c = 0; c < 2; ++c) {
    const __bf16* qp = Qh + (size_t)(q0 + l16) * HD + c * 32;
    qa[c] = frag_bf16(qp + 16 * hl, qp + 16 * hl + 8);
  }

  float mrow[8], lrow[8];
#pragma unroll
  for (int r = 0; r < 8; ++r) { mrow[r] = -3.0e38f; lrow[r] = 0.f; }
  v8f o[4];
#pragma unroll
  for (int i = 0; i < 4; ++i)
    o[i] = (v8f){0.f, 0.f, 0.f, 0.f, 0.f, 0.f, 0.f, 0.f};

  // cooperative stage of one 64-key K tile + V tile (128 threads x 8 chunks)
  auto stage_kv = [&](__bf16* ks, __bf16* vs, int t0) {
#pragma unroll
    for (int i = 0; i < 4; ++i) {
      int c = tid + i * 128;                 // 512 x 16B = K tile
      int row = c >> 3, seg = c & 7;
      async_copy16(Kh + (size_t)(t0 + row) * HD + seg * 8, ks + row * 72 + seg * 8);
    }
#pragma unroll
    for (int i = 0; i < 4; ++i) {
      int c = tid + i * 128;                 // 512 x 16B = V tile
      int row = c >> 3, seg = c & 7;
      async_copy16(Vh + (size_t)row * SEQ + t0 + seg * 8, vs + row * 72 + seg * 8);
    }
  };

  const int nIter = SEQ / 64;
  stage_kv(&Ks[0][0], &Vs[0][0], 0);

  for (int it = 0; it < nIter; ++it) {
    const int t0  = it * 64;
    const int cur = it & 1;
    wait_async();        // data for this iteration has landed in LDS
    __syncthreads();     // ...and every thread's compute(it-1) is finished
    if (it + 1 < nIter)  // prefetch next tile while we compute this one
      stage_kv(&Ks[cur ^ 1][0], &Vs[cur ^ 1][0], t0 + 64);

    const __bf16* ksc = &Ks[cur][0];
    const __bf16* vsc = &Vs[cur][0];

    // --- scores: S[16q x 64t] = Q * K^T; preload all 8 K fragments ---
    v8f s[4];
    {
      v16bf kf[4][2];
#pragma unroll
      for (int nt = 0; nt < 4; ++nt)
#pragma unroll
        for (int cc = 0; cc < 2; ++cc) {
          const __bf16* kp = ksc + (nt * 16 + l16) * 72 + cc * 32 + 16 * hl;
          kf[nt][cc] = frag_bf16(kp, kp + 8);
        }
#pragma unroll
      for (int nt = 0; nt < 4; ++nt) {
        v8f c = (v8f){0.f, 0.f, 0.f, 0.f, 0.f, 0.f, 0.f, 0.f};
        c = wmma_bf16(qa[0], kf[nt][0], c);
        c = wmma_bf16(qa[1], kf[nt][1], c);
        s[nt] = c;
      }
    }
    // --- online softmax: per-row reductions across the 16-lane N group ---
#pragma unroll
    for (int r = 0; r < 8; ++r) {
      float mx = fmaxf(fmaxf(s[0][r], s[1][r]), fmaxf(s[2][r], s[3][r]));
      mx = fmaxf(mx, __shfl_xor(mx, 1, 32));
      mx = fmaxf(mx, __shfl_xor(mx, 2, 32));
      mx = fmaxf(mx, __shfl_xor(mx, 4, 32));
      mx = fmaxf(mx, __shfl_xor(mx, 8, 32));
      const float mnew  = fmaxf(mrow[r], mx);
      const float alpha = __expf(mrow[r] - mnew);
      mrow[r] = mnew;
      const int row = r + 8 * hl;  // C-layout row for this VGPR
      float ps = 0.f;
#pragma unroll
      for (int nt = 0; nt < 4; ++nt) {
        float p = __expf(s[nt][r] - mnew);
        ps += p;
        pl[row * 72 + nt * 16 + l16] = (__bf16)p;  // C-layout -> LDS
      }
      ps += __shfl_xor(ps, 1, 32);
      ps += __shfl_xor(ps, 2, 32);
      ps += __shfl_xor(ps, 4, 32);
      ps += __shfl_xor(ps, 8, 32);
      lrow[r] = lrow[r] * alpha + ps;
#pragma unroll
      for (int dt = 0; dt < 4; ++dt) o[dt][r] *= alpha;
    }
    // P layout swap (C-layout stores -> A-layout reads, same wave; DS in-order)
    asm volatile("s_wait_dscnt 0" ::: "memory");
    // --- O += P * V; preload V fragments per K-chunk ---
#pragma unroll
    for (int cc = 0; cc < 2; ++cc) {
      const __bf16* pp = pl + l16 * 72 + cc * 32;
      v16bf pa = frag_bf16(pp + 16 * hl, pp + 16 * hl + 8);
      v16bf vf[4];
#pragma unroll
      for (int dt = 0; dt < 4; ++dt) {
        const __bf16* vp = vsc + (dt * 16 + l16) * 72 + cc * 32 + 16 * hl;
        vf[dt] = frag_bf16(vp, vp + 8);
      }
#pragma unroll
      for (int dt = 0; dt < 4; ++dt) o[dt] = wmma_bf16(pa, vf[dt], o[dt]);
    }
  }
#pragma unroll
  for (int r = 0; r < 8; ++r) {
    const float inv = 1.0f / lrow[r];
    const int m = q0 + r + 8 * hl;
#pragma unroll
    for (int dt = 0; dt < 4; ++dt)
      Abf[(size_t)m * HID + h * HD + dt * 16 + l16] = (__bf16)(o[dt][r] * inv);
  }
}

// ---------------------------------------------------------------------------
// Kernel 3: out = attn @ Wo + bo (fp32 out). grid (16, 12), block 256.
// Wave tile 32x64, B fragments preloaded and reused across both sub-tiles.
// ---------------------------------------------------------------------------
__global__ __launch_bounds__(256) void out_proj_kernel(
    const __bf16* __restrict__ Abf, const float* __restrict__ Wo,
    const float* __restrict__ bo, float* __restrict__ out) {
  __shared__ __bf16 Bs[64 * 72];
  const int m0 = blockIdx.x * 256;
  const int n0 = blockIdx.y * 64;
  const int tid  = threadIdx.x;
  const int lane = tid & 31;
  const int wv   = tid >> 5;
  const int hl   = lane >> 4;
  const int l16  = lane & 15;
  const int mrow0 = m0 + wv * 32;

  v8f acc[2][4];
#pragma unroll
  for (int s = 0; s < 2; ++s)
#pragma unroll
    for (int i = 0; i < 4; ++i)
      acc[s][i] = (v8f){0.f, 0.f, 0.f, 0.f, 0.f, 0.f, 0.f, 0.f};

  for (int kk = 0; kk < HID; kk += 32) {
    __syncthreads();
#pragma unroll
    for (int i = 0; i < 2; ++i) {
      int idx = tid + i * 256;
      int k = idx >> 4, ng = idx & 15;
      float4 f = *(const float4*)&Wo[(size_t)(kk + k) * HID + n0 + ng * 4];
      Bs[(ng * 4 + 0) * 72 + k] = (__bf16)f.x;
      Bs[(ng * 4 + 1) * 72 + k] = (__bf16)f.y;
      Bs[(ng * 4 + 2) * 72 + k] = (__bf16)f.z;
      Bs[(ng * 4 + 3) * 72 + k] = (__bf16)f.w;
    }
    __syncthreads();
    v16bf bfr[4];
#pragma unroll
    for (int nt = 0; nt < 4; ++nt) {
      const __bf16* bp = &Bs[(nt * 16 + l16) * 72 + 16 * hl];
      bfr[nt] = frag_bf16(bp, bp + 8);
    }
#pragma unroll
    for (int sub = 0; sub < 2; ++sub) {
      const __bf16* ap = Abf + (size_t)(mrow0 + sub * 16 + l16) * HID + kk;
      v16bf a = frag_bf16(ap + 16 * hl, ap + 16 * hl + 8);
#pragma unroll
      for (int nt = 0; nt < 4; ++nt)
        acc[sub][nt] = wmma_bf16(a, bfr[nt], acc[sub][nt]);
    }
  }

#pragma unroll
  for (int sub = 0; sub < 2; ++sub)
#pragma unroll
    for (int nt = 0; nt < 4; ++nt) {
      const int n = n0 + nt * 16 + l16;
      const float bias = bo[n];
#pragma unroll
      for (int r = 0; r < 8; ++r) {
        const int m = mrow0 + sub * 16 + r + 8 * hl;
        out[(size_t)m * HID + n] = acc[sub][nt][r] + bias;
      }
    }
}

extern "C" void kernel_launch(void* const* d_in, const int* in_sizes, int n_in,
                              void* d_out, int out_size, void* d_ws,
                              size_t ws_size, hipStream_t stream) {
  const float* x  = (const float*)d_in[0];
  const float* Wq = (const float*)d_in[1];
  const float* Wk = (const float*)d_in[2];
  const float* Wv = (const float*)d_in[3];
  const float* Wo = (const float*)d_in[4];
  const float* bo = (const float*)d_in[5];
  float* out = (float*)d_out;

  char* ws = (char*)d_ws;
  const size_t QSZ = (size_t)NH * SEQ * HD * sizeof(__bf16);  // 6 MB each
  __bf16* Qbf  = (__bf16*)(ws);
  __bf16* Kbf  = (__bf16*)(ws + QSZ);
  __bf16* Vtbf = (__bf16*)(ws + 2 * QSZ);
  __bf16* Abf  = (__bf16*)(ws + 3 * QSZ);  // [S][768] attn output, bf16

  qkv_proj_kernel<<<dim3(SEQ / 256, HID / 64, 3), 256, 0, stream>>>(
      x, Wq, Wk, Wv, Qbf, Kbf, Vtbf);
  flash_attn_kernel<<<dim3(NH * 64), 128, 0, stream>>>(Qbf, Kbf, Vtbf, Abf);
  out_proj_kernel<<<dim3(SEQ / 256, HID / 64), 256, 0, stream>>>(Abf, Wo, bo, out);

  (void)in_sizes; (void)n_in; (void)out_size; (void)ws_size;
}
